// LightGCN_31147102830630
// MI455X (gfx1250) — matrix operand order repaired and checked by reading, not compiled, not run on hardware
//
#include <hip/hip_runtime.h>
#include <cstdint>
#include <cstddef>

#define D 64
#define NLAYERS 3
#define WDECAY 1e-4f

typedef __attribute__((ext_vector_type(16))) _Float16 v16h;
typedef __attribute__((ext_vector_type(8)))  float    v8f;

// cur = acc = concat(user_emb, item_emb)
__global__ void lg_init(const float* __restrict__ ue, const float* __restrict__ ie,
                        float* __restrict__ cur, float* __restrict__ acc,
                        long nuE, long nE) {
  long i = ((long)blockIdx.x * blockDim.x + threadIdx.x) * 4;
  if (i >= nE) return;
  float4 v;
  if (i < nuE) v = *(const float4*)(ue + i);
  else         v = *(const float4*)(ie + (i - nuE));
  *(float4*)(cur + i) = v;
  *(float4*)(acc + i) = v;
}

__global__ void lg_zero4(float* __restrict__ p, long nE) {
  long i = ((long)blockIdx.x * blockDim.x + threadIdx.x) * 4;
  if (i >= nE) return;
  *(float4*)(p + i) = make_float4(0.f, 0.f, 0.f, 0.f);
}

__global__ void lg_zero_red(float* red) { red[0] = 0.f; red[1] = 0.f; }

// nxt[row,:] += val * cur[col,:]  — 16 lanes/edge, float4/lane, L2-resident scatter
__global__ void lg_spmm(const int* __restrict__ rows, const int* __restrict__ cols,
                        const float* __restrict__ vals,
                        const float* __restrict__ cur, float* __restrict__ nxt,
                        long nnz) {
  long t = (long)blockIdx.x * blockDim.x + threadIdx.x;
  int q = (int)(t & 15);
  long e = t >> 4;
  long stride = ((long)gridDim.x * blockDim.x) >> 4;
  for (; e < nnz; e += stride) {
    int r = rows[e];
    int c = cols[e];
    float v = vals[e];
    float4 x = *(const float4*)(cur + (size_t)c * D + q * 4);
    float* o = nxt + (size_t)r * D + q * 4;
    atomicAdd(o + 0, v * x.x);
    atomicAdd(o + 1, v * x.y);
    atomicAdd(o + 2, v * x.z);
    atomicAdd(o + 3, v * x.w);
  }
}

__global__ void lg_accadd(float* __restrict__ acc, const float* __restrict__ nxt, long nE) {
  long i = ((long)blockIdx.x * blockDim.x + threadIdx.x) * 4;
  if (i >= nE) return;
  float4 a = *(float4*)(acc + i);
  float4 b = *(const float4*)(nxt + i);
  a.x += b.x; a.y += b.y; a.z += b.z; a.w += b.w;
  *(float4*)(acc + i) = a;
}

// L2 regularization: sum of squares of gathered original embeddings -> red[1]
__global__ void lg_reg(const float* __restrict__ ue, const float* __restrict__ ie,
                       const int* __restrict__ users, const int* __restrict__ pos,
                       const int* __restrict__ neg, int batch, float* __restrict__ red) {
  int t = blockIdx.x * blockDim.x + threadIdx.x;
  float s = 0.f;
  if (t < batch * D) {
    int i = t >> 6, d = t & 63;
    float a = ue[(size_t)users[i] * D + d];
    float b = ie[(size_t)pos[i] * D + d];
    float c = ie[(size_t)neg[i] * D + d];
    s = a * a + b * b + c * c;
  }
  for (int off = 16; off > 0; off >>= 1) s += __shfl_down(s, off, 32);
  if ((threadIdx.x & 31) == 0) atomicAdd(red + 1, s);
}

// BPR dots via WMMA: per 16-row tile, D = U(16x64) * diff^T(64x16), keep diagonal.
// A layout (16-bit 16x32): lane m=lane&15, half=lane>>4; v0..3 -> K = kb+half*8+2v(+1);
//                          v4..7 -> K = kb+16+half*8+2(v-4)(+1).
// B layout (16-bit 32x16): col N=lane&15; same K mapping as A (v0..3 -> kb+half*8+2v(+1); v4..7 -> kb+16+half*8+2(v-4)(+1)).
// C/D f32 16x16: lanes 0-15 {M=r,N=lane}; lanes 16-31 {M=r+8,N=lane-16}.
__global__ void lg_bpr_wmma(const float* __restrict__ acc,
                            const int* __restrict__ users, const int* __restrict__ pos,
                            const int* __restrict__ neg, int nu, int nTiles,
                            float* __restrict__ red) {
  int gwave = (blockIdx.x * blockDim.x + threadIdx.x) >> 5;
  int lane = threadIdx.x & 31;
  if (gwave >= nTiles) return;  // wave-uniform; EXEC stays all-ones below
  int m = lane & 15;
  int half = lane >> 4;
  int bi = gwave * 16 + m;
  const float* urow = acc + (size_t)users[bi] * D;
  const float* prow = acc + ((size_t)nu + (size_t)pos[bi]) * D;
  const float* nrow = acc + ((size_t)nu + (size_t)neg[bi]) * D;

  v8f c = {};
#pragma unroll
  for (int kb = 0; kb < 64; kb += 32) {
    v16h A, B;
#pragma unroll
    for (int v = 0; v < 4; ++v) {
      int k0 = kb + half * 8 + 2 * v;
      A[2 * v]       = (_Float16)urow[k0];
      A[2 * v + 1]   = (_Float16)urow[k0 + 1];
      int k1 = kb + 16 + half * 8 + 2 * v;
      A[2 * (v + 4)]     = (_Float16)urow[k1];
      A[2 * (v + 4) + 1] = (_Float16)urow[k1 + 1];
    }
#pragma unroll
    for (int v = 0; v < 8; ++v) {
      int k = kb + (v < 4 ? half * 8 + 2 * v : 16 + half * 8 + 2 * (v - 4));
      B[2 * v]     = (_Float16)(nrow[k] - prow[k]);
      B[2 * v + 1] = (_Float16)(nrow[k + 1] - prow[k + 1]);
    }
    c = __builtin_amdgcn_wmma_f32_16x16x32_f16(false, A, false, B, (short)0, c, false, false);
  }

  // diagonal: lanes 0-7 -> c[lane]; lanes 24-31 -> c[lane-24]
  float x = 0.f;
  bool valid = false;
  if (lane < 8)        { x = c[lane];      valid = true; }
  else if (lane >= 24) { x = c[lane - 24]; valid = true; }
  x *= (1.0f / 16.0f);  // light_out = acc/4 on both sides of the dot
  float sp = valid ? (fmaxf(x, 0.f) + log1pf(expf(-fabsf(x)))) : 0.f;
  for (int off = 16; off > 0; off >>= 1) sp += __shfl_down(sp, off, 32);
  if (lane == 0) atomicAdd(red + 0, sp);
}

// scalar fallback for batch % 16 remainder
__global__ void lg_bpr_rem(const float* __restrict__ acc, const int* __restrict__ users,
                           const int* __restrict__ pos, const int* __restrict__ neg,
                           int nu, int start, int batch, float* __restrict__ red) {
  int i = start + blockIdx.x * blockDim.x + threadIdx.x;
  if (i >= batch) return;
  const float* urow = acc + (size_t)users[i] * D;
  const float* prow = acc + ((size_t)nu + (size_t)pos[i]) * D;
  const float* nrow = acc + ((size_t)nu + (size_t)neg[i]) * D;
  float x = 0.f;
  for (int k = 0; k < D; ++k) x += urow[k] * (nrow[k] - prow[k]);
  x *= (1.0f / 16.0f);
  atomicAdd(red, fmaxf(x, 0.f) + log1pf(expf(-fabsf(x))));
}

__global__ void lg_finalize(const float* __restrict__ red, float* __restrict__ out, float invB) {
  out[0] = red[0] * invB + WDECAY * 0.5f * red[1] * invB;
}

extern "C" void kernel_launch(void* const* d_in, const int* in_sizes, int n_in,
                              void* d_out, int out_size, void* d_ws, size_t ws_size,
                              hipStream_t stream) {
  const float* ue   = (const float*)d_in[0];
  const float* ie   = (const float*)d_in[1];
  const int*   rows = (const int*)d_in[2];
  const int*   cols = (const int*)d_in[3];
  const float* vals = (const float*)d_in[4];
  const int*   users = (const int*)d_in[5];
  const int*   pos   = (const int*)d_in[6];
  const int*   neg   = (const int*)d_in[7];

  long nuE  = in_sizes[0];         // NUM_USERS * 64
  long nE   = nuE + in_sizes[1];   // N * 64
  int  nu   = (int)(nuE / D);
  long nnz  = in_sizes[2];
  int  batch = in_sizes[5];

  float* cur = (float*)d_ws;
  float* nxt = cur + nE;
  float* acc = nxt + nE;
  float* red = acc + nE;           // red[0]=softplus sum, red[1]=sq sum

  const int tpb = 256;
  long vecBlocks = (nE / 4 + tpb - 1) / tpb;

  lg_init<<<vecBlocks, tpb, 0, stream>>>(ue, ie, cur, acc, nuE, nE);
  lg_zero_red<<<1, 1, 0, stream>>>(red);

  for (int l = 0; l < NLAYERS; ++l) {
    lg_zero4<<<vecBlocks, tpb, 0, stream>>>(nxt, nE);
    lg_spmm<<<2048, tpb, 0, stream>>>(rows, cols, vals, cur, nxt, nnz);
    lg_accadd<<<vecBlocks, tpb, 0, stream>>>(acc, nxt, nE);
    float* t = cur; cur = nxt; nxt = t;
  }

  int regBlocks = (batch * D + tpb - 1) / tpb;
  lg_reg<<<regBlocks, tpb, 0, stream>>>(ue, ie, users, pos, neg, batch, red);

  int nTiles = batch / 16;
  if (nTiles > 0) {
    int wavesPerBlock = tpb / 32;
    int bprBlocks = (nTiles + wavesPerBlock - 1) / wavesPerBlock;
    lg_bpr_wmma<<<bprBlocks, tpb, 0, stream>>>(acc, users, pos, neg, nu, nTiles, red);
  }
  int rem = batch - nTiles * 16;
  if (rem > 0)
    lg_bpr_rem<<<1, 64, 0, stream>>>(acc, users, pos, neg, nu, nTiles * 16, batch, red);

  lg_finalize<<<1, 1, 0, stream>>>(red, (float*)d_out, 1.0f / (float)batch);
}
